// _HashBlock_3083786518784
// MI455X (gfx1250) — matrix operand order repaired and checked by reading, not compiled, not run on hardware
//
#include <hip/hip_runtime.h>
#include <hip/hip_bf16.h>
#include <math.h>

// Problem constants from the reference
#define NF   128    // features
#define ND   16     // neighbors per node
#define NK   8      // hashers
#define NH   256    // hash dim
#define NO   256    // output dim
#define KW   2048   // K*H
#define TILE 16     // node rows per block

typedef __attribute__((ext_vector_type(16))) __bf16 v16bf;
typedef __attribute__((ext_vector_type(8)))  __bf16 v8bf;
typedef __attribute__((ext_vector_type(8)))  float  v8f;

// ---------------------------------------------------------------------------
// Kernel 1: fold the sign-hash + linear layers into two effective 128x256
// matrices and store them pre-swizzled for the WMMA B-matrix layout:
//   Bp[(chunk*256 + o)*32 + (kk & 31)]  where kk in [0,256):
//     kk <  128 : Mn[f=kk][o]      (neigh path, Rcat @ Wn.T)
//     kk >= 128 : Ms[f=kk-128][o]  (self  path, Rcat @ Ws.T)
// ---------------------------------------------------------------------------
__global__ void __launch_bounds__(128)
pack_B(const float* __restrict__ R,   // [K, F, H]
       const float* __restrict__ Ws,  // [O, K*H]
       const float* __restrict__ Wn,  // [O, K*H]
       __bf16* __restrict__ Bp)       // [8][256][32] bf16
{
    const int o = blockIdx.x;    // 0..255 (shared by whole block -> scalar W loads)
    const int f = threadIdx.x;   // 0..127

    const float* __restrict__ wn = Wn + (size_t)o * KW;
    const float* __restrict__ ws = Ws + (size_t)o * KW;

    float mn = 0.f, ms = 0.f;
    for (int k = 0; k < NK; ++k) {
        const float* __restrict__ r   = R  + ((size_t)k * NF + f) * NH;
        const float* __restrict__ wnk = wn + k * NH;
        const float* __restrict__ wsk = ws + k * NH;
        #pragma unroll 8
        for (int h = 0; h < NH; ++h) {
            float rv = r[h];
            mn = fmaf(rv, wnk[h], mn);
            ms = fmaf(rv, wsk[h], ms);
        }
    }

    // neigh path -> kk = f
    {
        int c = f >> 5, j = f & 31;
        Bp[((c * NO + o) << 5) + j] = (__bf16)mn;
    }
    // self path -> kk = 128 + f
    {
        int kk = NF + f;
        int c = kk >> 5, j = kk & 31;
        Bp[((c * NO + o) << 5) + j] = (__bf16)ms;
    }
}

// ---------------------------------------------------------------------------
// Kernel 2: per 16-node tile:
//   phase 1: build A = [mean_{d} x[nbr[n,d]] | x[n]] as bf16 in LDS (16 x 256)
//   phase 2: 4 waves x 4 column tiles; acc[16x16] += A(16x32) * B(32x16)
//            via v_wmma_f32_16x16x32_bf16 over 8 K-chunks; bias + ELU epilogue
// ---------------------------------------------------------------------------
#define A_PITCH 264   // 528 B/row -> rotates banks by 4 per row (conflict-free)

__global__ void __launch_bounds__(128)
hash_main(const float* __restrict__ x,     // [N, F]
          const int*   __restrict__ nbr,   // [N, D]
          const float* __restrict__ bias,  // [O]
          const __bf16* __restrict__ Bp,   // [8][256][32]
          float* __restrict__ out)         // [N, O]
{
    __shared__ __bf16 A[TILE][A_PITCH];

    const int tid   = threadIdx.x;          // 0..127
    const int node0 = blockIdx.x * TILE;

    // ---- Phase 1: gather + degree-mean, fill A tile ----
    const int f = tid;                      // one feature column per thread
    for (int m = 0; m < TILE; ++m) {
        const int node = node0 + m;
        const int* __restrict__ ni = nbr + (size_t)node * ND;   // uniform -> SMEM loads
        float s = 0.f;
        #pragma unroll
        for (int d = 0; d < ND; ++d) {
            s += x[(size_t)ni[d] * NF + f];  // coalesced 512B row gathers (L2-resident)
        }
        A[m][f]      = (__bf16)(s * (1.0f / ND));            // neigh_mean, kk = f
        A[m][NF + f] = (__bf16)x[(size_t)node * NF + f];     // self,       kk = 128+f
    }
    __syncthreads();

    // ---- Phase 2: WMMA ----
    const int wave = tid >> 5;
    const int lane = tid & 31;
    const int mrow = lane & 15;              // A row this lane feeds
    const int kbA  = (lane >> 4) * 16;       // A k-offset inside a 32-chunk
    const int kbB  = (lane >> 4) * 16;       // B k-offset inside a 32-chunk
    const int mb   = (lane >> 4) * 8;        // first output row held by this lane

    #pragma unroll
    for (int i = 0; i < 4; ++i) {
        const int ot = wave * 4 + i;         // column tile 0..15
        const int o  = ot * 16 + (lane & 15);

        v8f acc = {};
        #pragma unroll
        for (int c = 0; c < 8; ++c) {
            // A fragment: two 16B LDS loads (K=kb..kb+7 and K=kb+8..kb+15 of chunk c)
            v8bf alo = *(const v8bf*)&A[mrow][c * 32 + kbA];
            v8bf ahi = *(const v8bf*)&A[mrow][c * 32 + 8 + kbA];
            v16bf av = __builtin_shufflevector(alo, ahi,
                0, 1, 2, 3, 4, 5, 6, 7, 8, 9, 10, 11, 12, 13, 14, 15);
            // B fragment: one contiguous 32B global load (pre-swizzled)
            v16bf bv = *(const v16bf*)&Bp[(((size_t)c * NO + o) << 5) + kbB];

            acc = __builtin_amdgcn_wmma_f32_16x16x32_bf16(
                /*neg_a=*/false, av, /*neg_b=*/false, bv,
                /*c_mod=*/(short)0, acc, /*reuse_a=*/false, /*reuse_b=*/false);
        }

        // Epilogue: bias + ELU(alpha=1), scatter per C/D layout
        const float b = bias[o];
        #pragma unroll
        for (int r = 0; r < 8; ++r) {
            float z = acc[r] + b;
            z = (z > 0.f) ? z : expm1f(z);
            out[(size_t)(node0 + mb + r) * NO + o] = z;
        }
    }
}

// ---------------------------------------------------------------------------
extern "C" void kernel_launch(void* const* d_in, const int* in_sizes, int n_in,
                              void* d_out, int out_size, void* d_ws, size_t ws_size,
                              hipStream_t stream)
{
    const float* x    = (const float*)d_in[0];  // [N, 128]
    const int*   nbr  = (const int*)  d_in[1];  // [N, 16]
    const float* R    = (const float*)d_in[2];  // [8, 128, 256]
    const float* Ws   = (const float*)d_in[3];  // [256, 2048]
    const float* Wn   = (const float*)d_in[4];  // [256, 2048]
    const float* bias = (const float*)d_in[5];  // [256]
    float* out = (float*)d_out;

    __bf16* Bp = (__bf16*)d_ws;                 // 8*256*32 bf16 = 128 KB

    const int N = in_sizes[0] / NF;             // 100000

    pack_B<<<NO, 128, 0, stream>>>(R, Ws, Wn, Bp);
    hash_main<<<N / TILE, 128, 0, stream>>>(x, nbr, bias, Bp, out);
}
